// RowEncoder_28132035789450
// MI455X (gfx1250) — compile-verified, matrix operand order checked by reading, not run on hardware
//
#include <hip/hip_runtime.h>
#include <hip/hip_bf16.h>

// ---------------------------------------------------------------------------
// Transformer row-encoder forward for gfx1250 (MI455X), WMMA f16 -> f32,
// with async global->LDS tile staging (GLOBAL_LOAD_ASYNC_TO_LDS_B128).
// Shapes: S=512 sequences, L=132 tokens (pad 144/160), D=512, H=8, Dh=64, F=2048
// ---------------------------------------------------------------------------

typedef __attribute__((ext_vector_type(16))) _Float16 v16h;
typedef __attribute__((ext_vector_type(8)))  _Float16 v8h;
typedef __attribute__((ext_vector_type(8)))  float    v8f;

#define M_TOT   67584        // 512 * 132
#define D_MOD   512
#define D_FF    2048
#define SEQS    512
#define NHEAD   8
#define LPAD    144          // 9 * 16  (row pad of L=132)
#define LKPAD   160          // 5 * 32  (K-dim pad for AV)

// ---- WMMA helpers ---------------------------------------------------------

__device__ __forceinline__ v8f wmma_f16(v16h a, v16h b, v8f c) {
  // D = A(16x32, f16) x B(32x16, f16) + C(16x16, f32)
  return __builtin_amdgcn_wmma_f32_16x16x32_f16(
      /*neg_a=*/false, a, /*neg_b=*/false, b,
      /*c_mod=*/(short)0, c, /*reuse_a=*/false, /*reuse_b=*/false);
}

// Load a 16x32 f16 fragment (A-style striping) from row-major memory.
// Per CDNA5 ISA 7.12.2: lane(0..15) holds row M=lane, halves map
// K = {0..7,16..23}; lanes 16..31 hold K = {8..15,24..31} of row lane-16.
// The same striping serves as the B (32x16) operand when `base` points at a
// [16 rows = N columns][32 = K] pre-transposed tile.
__device__ __forceinline__ v16h ld_frag(const _Float16* base, int ld) {
  const int lane = threadIdx.x & 31;
  const int r    = lane & 15;
  const int kb   = (lane & 16) ? 8 : 0;
  const _Float16* p = base + r * ld + kb;
  v8h lo = *(const v8h*)(p);
  v8h hi = *(const v8h*)(p + 16);
  v16h o;
#pragma unroll
  for (int i = 0; i < 8; ++i) { o[i] = lo[i]; o[i + 8] = hi[i]; }
  return o;
}

// Generic pointer -> raw LDS byte offset (ISA 10.2: LDS address is the low 32 bits).
__device__ __forceinline__ unsigned lds_addr(const void* p) {
  return (unsigned)(uintptr_t)p;
}

// ---- Elementwise / setup kernels ------------------------------------------

__global__ __launch_bounds__(256)
void k_build_h(const float* __restrict__ X, const float* __restrict__ CLS,
               float* __restrict__ h32) {
  size_t idx = (size_t)blockIdx.x * 256 + threadIdx.x;   // m*512 + d
  int d = (int)(idx & 511);
  size_t m = idx >> 9;
  int l = (int)(m % 132);
  int s = (int)(m / 132);
  float v = (l < 4) ? CLS[l * 512 + d]
                    : X[((size_t)s * 128 + (l - 4)) * 512 + d];
  h32[idx] = v;
}

__global__ __launch_bounds__(256)
void k_cvt(const float* __restrict__ x, _Float16* __restrict__ y, size_t n) {
  size_t i = (size_t)blockIdx.x * 256 + threadIdx.x;
  if (i < n) y[i] = (_Float16)x[i];
}

// W[K,N] f32 -> Wt[N,K] f16 (so B fragments load contiguously).
__global__ __launch_bounds__(256)
void k_pack_wt(const float* __restrict__ w, _Float16* __restrict__ wt,
               int K, int N) {
  size_t i = (size_t)blockIdx.x * 256 + threadIdx.x;
  if (i >= (size_t)K * N) return;
  int n = (int)(i / K), k = (int)(i % K);
  wt[i] = (_Float16)w[(size_t)k * N + n];
}

// ---- Tiled GEMM: C[M,N] = A[M,K](f16) * Wt[N,K]^T(f16) + bias -------------
// 128 threads = 4 waves; workgroup tile 64x64; each wave computes 16x64.
// Double-buffered ASYNC global->LDS staging: tiles bypass VGPRs, tracked by
// ASYNCcnt; s_wait_asynccnt 4 keeps exactly the next stage's 4 ops in flight.
// EPI: 0 = f32 out, 1 = f16 out, 2 = f16 out with exact GELU.
template <int EPI>
__global__ __launch_bounds__(128)
void k_gemm(const _Float16* __restrict__ A, const _Float16* __restrict__ Wt,
            const float* __restrict__ bias, void* __restrict__ Cout,
            int N, int K) {
  __shared__ _Float16 As[2][64 * 32];
  __shared__ _Float16 Bs[2][64 * 32];
  const int tid  = threadIdx.x;
  const int wave = tid >> 5;
  const int lane = tid & 31;
  const size_t m0 = (size_t)blockIdx.y * 64;
  const int    n0 = blockIdx.x * 64;

  v8f acc[4] = {};

  const int lr = tid >> 1;          // 0..63 : tile row
  const int lc = (tid & 1) * 16;    // 0 / 16: half-offset within 32-wide K slab

  const _Float16* gA = A  + (m0 + lr) * (size_t)K + lc;
  const _Float16* gB = Wt + (size_t)(n0 + lr) * (size_t)K + lc;

  // Issue one stage: 4 async B128 per wave (A lo/hi, B lo/hi halves).
  auto issue = [&](int buf, int k0) {
    unsigned la  = lds_addr(&As[buf][lr * 32 + lc]);
    unsigned lb  = lds_addr(&Bs[buf][lr * 32 + lc]);
    unsigned long long ga = (unsigned long long)(gA + k0);
    unsigned long long gb = (unsigned long long)(gB + k0);
    asm volatile(
        "global_load_async_to_lds_b128 %0, %4, off\n\t"
        "global_load_async_to_lds_b128 %1, %4, off offset:16\n\t"
        "global_load_async_to_lds_b128 %2, %5, off\n\t"
        "global_load_async_to_lds_b128 %3, %5, off offset:16"
        :
        : "v"(la), "v"(la + 16), "v"(lb), "v"(lb + 16), "v"(ga), "v"(gb)
        : "memory");
  };

  const int nsteps = K >> 5;
  issue(0, 0);
  for (int i = 0; i < nsteps; ++i) {
    if (i + 1 < nsteps) {
      issue((i + 1) & 1, (i + 1) * 32);
      asm volatile("s_wait_asynccnt 0x4" ::: "memory");  // stage i landed
    } else {
      asm volatile("s_wait_asynccnt 0x0" ::: "memory");
    }
    __syncthreads();                                      // all waves see LDS

    const _Float16* as = As[i & 1];
    const _Float16* bs = Bs[i & 1];
    v16h af = ld_frag(as + wave * 16 * 32, 32);
#pragma unroll
    for (int ct = 0; ct < 4; ++ct) {
      v16h bf = ld_frag(bs + ct * 16 * 32, 32);
      acc[ct] = wmma_f16(af, bf, acc[ct]);
    }
    __syncthreads();   // done reading buf before it is refilled next+1 stage
  }

  const int hi8  = (lane & 16) ? 8 : 0;
  const int lcol = lane & 15;
#pragma unroll
  for (int ct = 0; ct < 4; ++ct) {
    const int col = n0 + ct * 16 + lcol;
    const float bv = bias[col];
#pragma unroll
    for (int j = 0; j < 8; ++j) {
      const size_t row = m0 + wave * 16 + j + hi8;
      float v = acc[ct][j] + bv;
      if (EPI == 2) v = 0.5f * v * (1.0f + erff(v * 0.70710678118f));
      if (EPI == 0) ((float*)Cout)[row * (size_t)N + col] = v;
      else          ((_Float16*)Cout)[row * (size_t)N + col] = (_Float16)v;
    }
  }
}

// ---- RoPE + padded head packing -------------------------------------------
// q16/k16: [M][512] f16 -> qh/kh: [S*H][LPAD][64] f16 (rows >= 132 zeroed).
__global__ __launch_bounds__(256)
void k_rope_qk(const _Float16* __restrict__ q16, const _Float16* __restrict__ k16,
               _Float16* __restrict__ qh, _Float16* __restrict__ kh) {
  size_t idx = (size_t)blockIdx.x * 256 + threadIdx.x;  // ((s*144+lp)*8+h)*32+d
  int d = (int)(idx & 31);
  int h = (int)((idx >> 5) & 7);
  size_t r = idx >> 8;
  int lp = (int)(r % 144);
  int s  = (int)(r / 144);
  size_t o = ((size_t)(s * 8 + h) * LPAD + lp) * 64 + d;
  if (lp >= 132) {
    qh[o] = (_Float16)0.f; qh[o + 32] = (_Float16)0.f;
    kh[o] = (_Float16)0.f; kh[o + 32] = (_Float16)0.f;
    return;
  }
  size_t mi = ((size_t)s * 132 + lp) * 512 + h * 64 + d;
  float q1 = (float)q16[mi], q2 = (float)q16[mi + 32];
  float k1 = (float)k16[mi], k2 = (float)k16[mi + 32];
  // inv_freq = 100000^(-d/32) = exp(-d * ln(1e5)/32)
  float ang = (float)lp * __expf(-(float)d * 0.3597789188f);
  float sn, cs;
  __sincosf(ang, &sn, &cs);
  qh[o]      = (_Float16)(q1 * cs - q2 * sn);
  qh[o + 32] = (_Float16)(q2 * cs + q1 * sn);
  kh[o]      = (_Float16)(k1 * cs - k2 * sn);
  kh[o + 32] = (_Float16)(k2 * cs + k1 * sn);
}

// v16: [M][512] -> vt: [S*H][64][LKPAD] (transposed; cols >= 132 zeroed)
__global__ __launch_bounds__(256)
void k_pack_vt(const _Float16* __restrict__ v16, _Float16* __restrict__ vt) {
  size_t idx = (size_t)blockIdx.x * 256 + threadIdx.x;  // (hs*64+d)*160+j
  int j = (int)(idx % LKPAD);
  size_t r = idx / LKPAD;
  int d = (int)(r & 63);
  int hs = (int)(r >> 6);
  int s = hs >> 3, h = hs & 7;
  _Float16 val = (_Float16)0.f;
  if (j < 132) val = v16[((size_t)s * 132 + j) * 512 + h * 64 + d];
  vt[idx] = val;
}

// ---- Fused scores + softmax -----------------------------------------------
// One wave per (seq,head, 16-query-row tile): S = QK^T/8, masked softmax,
// writes P f16 [S*H][LPAD][LKPAD] with K-dim zero padding.
__global__ __launch_bounds__(128)
void k_scores(const _Float16* __restrict__ qh, const _Float16* __restrict__ kh,
              _Float16* __restrict__ p16) {
  const int wv = blockIdx.x * 4 + (threadIdx.x >> 5);
  const int qt = wv % 9;
  const int hs = wv / 9;
  const int lane = threadIdx.x & 31;

  const _Float16* Q  = qh + (size_t)hs * (LPAD * 64);
  const _Float16* Kb = kh + (size_t)hs * (LPAD * 64);

  v16h a0 = ld_frag(Q + qt * 16 * 64, 64);
  v16h a1 = ld_frag(Q + qt * 16 * 64 + 32, 64);

  v8f acc[9];
#pragma unroll
  for (int jt = 0; jt < 9; ++jt) {
    v16h b0 = ld_frag(Kb + jt * 16 * 64, 64);
    v16h b1 = ld_frag(Kb + jt * 16 * 64 + 32, 64);
    v8f c = {};
    c = wmma_f16(a0, b0, c);
    c = wmma_f16(a1, b1, c);
    acc[jt] = c;
  }

  const int lcol = lane & 15;
  const int hi8  = (lane & 16) ? 8 : 0;

  float mx[8], sm[8];
#pragma unroll
  for (int j = 0; j < 8; ++j) mx[j] = -3.0e38f;
#pragma unroll
  for (int jt = 0; jt < 9; ++jt) {
    const bool valid = (jt * 16 + lcol) < 132;
#pragma unroll
    for (int j = 0; j < 8; ++j) {
      float v = valid ? acc[jt][j] * 0.125f : -1.0e30f;  // 1/sqrt(64), mask
      acc[jt][j] = v;
      mx[j] = fmaxf(mx[j], v);
    }
  }
  // Row lives across one 16-lane half-wave: reduce with xor masks 1..8.
#pragma unroll
  for (int j = 0; j < 8; ++j) {
#pragma unroll
    for (int o = 1; o < 16; o <<= 1)
      mx[j] = fmaxf(mx[j], __shfl_xor(mx[j], o, 32));
    sm[j] = 0.0f;
  }
#pragma unroll
  for (int jt = 0; jt < 9; ++jt) {
#pragma unroll
    for (int j = 0; j < 8; ++j) {
      float e = __expf(acc[jt][j] - mx[j]);
      acc[jt][j] = e;
      sm[j] += e;
    }
  }
#pragma unroll
  for (int j = 0; j < 8; ++j) {
#pragma unroll
    for (int o = 1; o < 16; o <<= 1) sm[j] += __shfl_xor(sm[j], o, 32);
    sm[j] = 1.0f / sm[j];
  }

  _Float16* P = p16 + (size_t)hs * (LPAD * LKPAD);
#pragma unroll
  for (int j = 0; j < 8; ++j) {
    const int row = qt * 16 + j + hi8;
#pragma unroll
    for (int jt = 0; jt < 9; ++jt)
      P[(size_t)row * LKPAD + jt * 16 + lcol] = (_Float16)(acc[jt][j] * sm[j]);
  }
  // Zero K-pad columns [144,160) for this row tile.
  v8h z = {};
  const int zr = qt * 16 + lcol;
  *(v8h*)&P[(size_t)zr * LKPAD + 144 + hi8] = z;
}

// ---- AV: O = P x V (per head), heads concatenated into f16 A-operand ------
__global__ __launch_bounds__(128)
void k_av(const _Float16* __restrict__ p16, const _Float16* __restrict__ vt,
          _Float16* __restrict__ out16) {
  const int wv = blockIdx.x * 4 + (threadIdx.x >> 5);
  const int qt = wv % 9;
  const int hs = wv / 9;
  const int s = hs >> 3, h = hs & 7;
  const int lane = threadIdx.x & 31;

  const _Float16* P = p16 + (size_t)hs * (LPAD * LKPAD) + qt * 16 * LKPAD;
  const _Float16* V = vt + (size_t)hs * (64 * LKPAD);

  v8f acc[4] = {};
#pragma unroll
  for (int ks = 0; ks < 5; ++ks) {           // K = 160 = 5 * 32
    v16h af = ld_frag(P + ks * 32, LKPAD);
#pragma unroll
    for (int dt = 0; dt < 4; ++dt) {
      v16h bf = ld_frag(V + dt * 16 * LKPAD + ks * 32, LKPAD);
      acc[dt] = wmma_f16(af, bf, acc[dt]);
    }
  }
  const int lcol = lane & 15;
  const int hi8  = (lane & 16) ? 8 : 0;
#pragma unroll
  for (int j = 0; j < 8; ++j) {
    const int row = qt * 16 + j + hi8;
    if (row < 132) {
      const size_t m = (size_t)s * 132 + row;
#pragma unroll
      for (int dt = 0; dt < 4; ++dt)
        out16[m * 512 + h * 64 + dt * 16 + lcol] = (_Float16)acc[dt][j];
    }
  }
}

// ---- LayerNorm: one row per block -----------------------------------------
// mode 0: outA = LN(x + r)      (r may be null -> LN(x))
// mode 1: final LN, scattered into d_out (cls_out then feat_out)
__global__ __launch_bounds__(256)
void k_ln(const float* __restrict__ x, const float* __restrict__ r,
          const float* __restrict__ g, const float* __restrict__ b,
          float* __restrict__ outA, float* __restrict__ outF, int mode) {
  __shared__ float red[256];
  const size_t m = blockIdx.x;
  const int t = threadIdx.x;
  float v0 = x[m * 512 + t];
  float v1 = x[m * 512 + t + 256];
  if (r) { v0 += r[m * 512 + t]; v1 += r[m * 512 + t + 256]; }
  red[t] = v0 + v1;
  __syncthreads();
  for (int o = 128; o > 0; o >>= 1) {
    if (t < o) red[t] += red[t + o];
    __syncthreads();
  }
  const float mean = red[0] * (1.0f / 512.0f);
  __syncthreads();
  float d0 = v0 - mean, d1 = v1 - mean;
  red[t] = d0 * d0 + d1 * d1;
  __syncthreads();
  for (int o = 128; o > 0; o >>= 1) {
    if (t < o) red[t] += red[t + o];
    __syncthreads();
  }
  const float inv = rsqrtf(red[0] * (1.0f / 512.0f) + 1e-5f);
  const float y0 = d0 * inv * g[t] + b[t];
  const float y1 = d1 * inv * g[t + 256] + b[t + 256];
  if (mode == 0) {
    outA[m * 512 + t] = y0;
    outA[m * 512 + t + 256] = y1;
  } else {
    int s = (int)(m / 132), l = (int)(m % 132);
    size_t o0 = (l < 4)
        ? ((size_t)s * 2048 + (size_t)l * 512)
        : (size_t)1048576 + ((size_t)s * 128 + (l - 4)) * 512;
    outF[o0 + t] = y0;
    outF[o0 + t + 256] = y1;
  }
}

// ---------------------------------------------------------------------------
extern "C" void kernel_launch(void* const* d_in, const int* in_sizes, int n_in,
                              void* d_out, int out_size, void* d_ws, size_t ws_size,
                              hipStream_t stream) {
  (void)in_sizes; (void)n_in; (void)out_size; (void)ws_size;

  const size_t M = M_TOT;
  const size_t MD = M * 512;

  // ---- workspace carve-out (256B aligned) ----
  char* ws = (char*)d_ws;
  auto carve = [&](size_t bytes) {
    char* p = ws;
    ws += (bytes + 255) & ~(size_t)255;
    return p;
  };
  float*    h32 = (float*)carve(MD * 4);                    // residual stream
  float*    t32 = (float*)carve(MD * 4);                    // f32 GEMM out
  _Float16* a0  = (_Float16*)carve(MD * 2);                 // f16 A operand
  _Float16* ah  = (_Float16*)carve(M * 2048 * 2);           // QKV / FFN hidden
  _Float16* qhb = (_Float16*)carve((size_t)SEQS * NHEAD * LPAD * 64 * 2);
  _Float16* khb = (_Float16*)carve((size_t)SEQS * NHEAD * LPAD * 64 * 2);
  _Float16* vtb = (_Float16*)carve((size_t)SEQS * NHEAD * 64 * LKPAD * 2);
  _Float16* pb  = (_Float16*)carve((size_t)SEQS * NHEAD * LPAD * LKPAD * 2);
  _Float16* wt[3][6];
  const int wK[6] = {512, 512, 512, 512, 512, 2048};
  const int wN[6] = {512, 512, 512, 512, 2048, 512};
  for (int i = 0; i < 3; ++i)
    for (int w = 0; w < 6; ++w)
      wt[i][w] = (_Float16*)carve((size_t)wK[w] * wN[w] * 2);

  // ---- input indexing (setup_inputs dict order) ----
  const float* X   = (const float*)d_in[0];
  const float* CLS = (const float*)d_in[1];
  // per block (stride 16): wq.w,wq.b,wk.w,wk.b,wv.w,wv.b,wo.w,wo.b,
  //                        ln1_g,ln1_b,w1.w,w1.b,w2.w,w2.b,ln2_g,ln2_b
  const int wIdx[6] = {0, 2, 4, 6, 10, 12};  // weight matrices within block

  // ---- pack all weights (transpose + f16) ----
  for (int i = 0; i < 3; ++i) {
    const int base = 2 + 16 * i;
    for (int w = 0; w < 6; ++w) {
      size_t tot = (size_t)wK[w] * wN[w];
      k_pack_wt<<<(unsigned)((tot + 255) / 256), 256, 0, stream>>>(
          (const float*)d_in[base + wIdx[w]], wt[i][w], wK[w], wN[w]);
    }
  }

  // ---- build h0 = concat(cls, x) ----
  k_build_h<<<(unsigned)(MD / 256), 256, 0, stream>>>(X, CLS, h32);

  const dim3 gP(512 / 64, (unsigned)(M / 64));     // N=512 GEMMs
  const dim3 gF(2048 / 64, (unsigned)(M / 64));    // N=2048 GEMM
  const unsigned attnBlocks = SEQS * NHEAD * 9 / 4;

  for (int i = 0; i < 3; ++i) {
    const int base = 2 + 16 * i;
    const float* bq  = (const float*)d_in[base + 1];
    const float* bk  = (const float*)d_in[base + 3];
    const float* bv  = (const float*)d_in[base + 5];
    const float* bo  = (const float*)d_in[base + 7];
    const float* g1  = (const float*)d_in[base + 8];
    const float* be1 = (const float*)d_in[base + 9];
    const float* b1  = (const float*)d_in[base + 11];
    const float* b2  = (const float*)d_in[base + 13];
    const float* g2  = (const float*)d_in[base + 14];
    const float* be2 = (const float*)d_in[base + 15];

    // --- attention ---
    k_cvt<<<(unsigned)(MD / 256), 256, 0, stream>>>(h32, a0, MD);
    k_gemm<1><<<gP, 128, 0, stream>>>(a0, wt[i][0], bq, ah,            512, 512);
    k_gemm<1><<<gP, 128, 0, stream>>>(a0, wt[i][1], bk, ah + MD,       512, 512);
    k_gemm<1><<<gP, 128, 0, stream>>>(a0, wt[i][2], bv, ah + 2 * MD,   512, 512);
    k_rope_qk<<<(unsigned)((size_t)SEQS * LPAD * NHEAD * 32 / 256), 256, 0, stream>>>(
        ah, ah + MD, qhb, khb);
    k_pack_vt<<<(unsigned)((size_t)SEQS * NHEAD * 64 * LKPAD / 256), 256, 0, stream>>>(
        ah + 2 * MD, vtb);
    k_scores<<<attnBlocks, 128, 0, stream>>>(qhb, khb, pb);
    k_av<<<attnBlocks, 128, 0, stream>>>(pb, vtb, a0);
    k_gemm<0><<<gP, 128, 0, stream>>>(a0, wt[i][3], bo, t32, 512, 512);
    k_ln<<<(unsigned)M, 256, 0, stream>>>(h32, t32, g1, be1, h32, nullptr, 0);

    // --- FFN ---
    k_cvt<<<(unsigned)(MD / 256), 256, 0, stream>>>(h32, a0, MD);
    k_gemm<2><<<gF, 128, 0, stream>>>(a0, wt[i][4], b1, ah, 2048, 512);
    k_gemm<0><<<gP, 128, 0, stream>>>(ah, wt[i][5], b2, t32, 512, 2048);
    k_ln<<<(unsigned)M, 256, 0, stream>>>(h32, t32, g2, be2, h32, nullptr, 0);
  }

  // ---- final LN + scatter into d_out (cls_out then feat_out) ----
  const float* og = (const float*)d_in[50];
  const float* ob = (const float*)d_in[51];
  k_ln<<<(unsigned)M, 256, 0, stream>>>(h32, nullptr, og, ob, nullptr,
                                        (float*)d_out, 1);
}
